// TemporalGCNCell_57123065037360
// MI455X (gfx1250) — matrix operand order repaired and checked by reading, not compiled, not run on hardware
//
#include <hip/hip_runtime.h>
#include <math.h>

typedef __attribute__((ext_vector_type(16))) __bf16 v16bf;
typedef __attribute__((ext_vector_type(8)))  float  v8f;

#define HID 128

// ---------------------------------------------------------------------------
// Degree / normalization kernels
// ---------------------------------------------------------------------------
__global__ void k_init_deg(float* deg, int N) {
    int i = blockIdx.x * blockDim.x + threadIdx.x;
    if (i < N) deg[i] = 1.0f;                 // self-loop
}

__global__ void k_deg_edges(const int* __restrict__ dst, float* deg, int E) {
    int e = blockIdx.x * blockDim.x + threadIdx.x;
    if (e < E) atomicAdd(&deg[dst[e]], 1.0f);
}

__global__ void k_deg_rsqrt(float* deg, int N) {
    int i = blockIdx.x * blockDim.x + threadIdx.x;
    if (i < N) deg[i] = rsqrtf(deg[i]);       // deg >= 1
}

// ---------------------------------------------------------------------------
// Pack fp32 weight [K,128] row-major -> bf16 WMMA B-fragment layout.
// Element index: ((ks*8 + ct)*32 + lane)*16 + j
//   lane 0-15  : column N = ct*16+lane, K = ks*32 + j        (j = 0..15)
//   lane 16-31 : column N = ct*16+lane-16, K = ks*32 + 16 + j
// ---------------------------------------------------------------------------
__global__ void k_pack_w(const float* __restrict__ W, __bf16* __restrict__ out, int K) {
    int tid = blockIdx.x * blockDim.x + threadIdx.x;
    int total = K * HID;
    if (tid >= total) return;
    int j    = tid & 15;
    int lane = (tid >> 4) & 31;
    int ct   = (tid >> 9) & 7;
    int ks   = tid >> 12;
    int k = ks * 32 + ((lane & 16) ? 16 : 0) + j;
    int n = ct * 16 + (lane & 15);
    out[tid] = (__bf16)W[k * HID + n];
}

// ---------------------------------------------------------------------------
// WMMA GEMM:  C[N,128] = [A0 | A1][N,K] @ Bpacked[K,128]    (K = 128 or 256)
// A0 covers columns 0..127, A1 columns 128..255 (both row-major, stride 128).
// 256 threads = 8 waves; wave w computes 32 rows x 128 cols:
//   rows [blk*256 + 32w, +32)  as two 16-row A fragments -> 16 acc tiles.
// Each B fragment (loaded in groups of 4 to distinct regs) feeds 2 WMMAs.
// ---------------------------------------------------------------------------
__device__ __forceinline__ v16bf make_a_frag(const float* Ar) {
    float4 f0 = *(const float4*)(Ar);
    float4 f1 = *(const float4*)(Ar + 4);
    float4 f2 = *(const float4*)(Ar + 8);
    float4 f3 = *(const float4*)(Ar + 12);
    v16bf a;
    a[0]=(__bf16)f0.x;  a[1]=(__bf16)f0.y;  a[2]=(__bf16)f0.z;  a[3]=(__bf16)f0.w;
    a[4]=(__bf16)f1.x;  a[5]=(__bf16)f1.y;  a[6]=(__bf16)f1.z;  a[7]=(__bf16)f1.w;
    a[8]=(__bf16)f2.x;  a[9]=(__bf16)f2.y;  a[10]=(__bf16)f2.z; a[11]=(__bf16)f2.w;
    a[12]=(__bf16)f3.x; a[13]=(__bf16)f3.y; a[14]=(__bf16)f3.z; a[15]=(__bf16)f3.w;
    return a;
}

__global__ void __launch_bounds__(256)
k_gemm_bf16(const float* __restrict__ A0, const float* __restrict__ A1,
            const __bf16* __restrict__ Bp, float* __restrict__ C,
            int N, int K) {
    const int wave = threadIdx.x >> 5;
    const int lane = threadIdx.x & 31;
    const int half = lane >> 4;                 // 0 | 1
    const int l16  = lane & 15;
    const int rowBase = blockIdx.x * 256 + wave * 32;

    int r0 = rowBase + l16;       if (r0 > N - 1) r0 = N - 1;   // clamp loads
    int r1 = rowBase + 16 + l16;  if (r1 > N - 1) r1 = N - 1;

    v8f acc0[8] = {};   // rows rowBase..+15
    v8f acc1[8] = {};   // rows rowBase+16..+31

    const int nks = K >> 5;                     // k-steps of 32
    for (int ks = 0; ks < nks; ++ks) {
        const int kb = ks << 5;
        const float* base = (kb < 128) ? A0 : A1;
        const int kk = kb & 127;
        const int o0 = half ? 16 : 0;           // lanes 0-15: K 0-15; 16-31: K 16-31

        v16bf a0 = make_a_frag(base + (size_t)r0 * HID + kk + o0);
        v16bf a1 = make_a_frag(base + (size_t)r1 * HID + kk + o0);

        const __bf16* bp = Bp + ((size_t)ks * 256 + lane) * 16;
#pragma unroll
        for (int g = 0; g < 2; ++g) {           // two groups of 4 column tiles
            v16bf b[4];
#pragma unroll
            for (int q = 0; q < 4; ++q)
                b[q] = *(const v16bf*)(bp + (size_t)(g * 4 + q) * 512);
#pragma unroll
            for (int q = 0; q < 4; ++q) {
                const int ct = g * 4 + q;
                acc0[ct] = __builtin_amdgcn_wmma_f32_16x16x32_bf16(
                    false, a0, false, b[q], (short)0, acc0[ct], false, false);
                acc1[ct] = __builtin_amdgcn_wmma_f32_16x16x32_bf16(
                    false, a1, false, b[q], (short)0, acc1[ct], false, false);
            }
        }
    }

    // C/D layout: VGPR r -> M = r + 8*half; N = lane&15
#pragma unroll
    for (int ct = 0; ct < 8; ++ct) {
#pragma unroll
        for (int r = 0; r < 8; ++r) {
            int row0 = rowBase + r + half * 8;
            int row1 = row0 + 16;
            int col  = ct * 16 + l16;
            if (row0 < N) C[(size_t)row0 * HID + col] = acc0[ct][r];
            if (row1 < N) C[(size_t)row1 * HID + col] = acc1[ct][r];
        }
    }
}

// ---------------------------------------------------------------------------
// GCN aggregation helpers
// ---------------------------------------------------------------------------
// xs = xw * dinv[row]; agg initialized with self-loop contribution (= xs)
__global__ void k_scale_init(const float* __restrict__ xw, const float* __restrict__ dinv,
                             float* __restrict__ xs, float* __restrict__ agg, long long total) {
    long long i = (long long)blockIdx.x * blockDim.x + threadIdx.x;
    if (i >= total) return;
    float v = xw[i] * dinv[i >> 7];
    xs[i]  = v;
    agg[i] = v;
}

// agg[dst] += xs[src]  (4 features per thread, 32 threads per edge)
__global__ void k_scatter(const int* __restrict__ src, const int* __restrict__ dst,
                          const float* __restrict__ xs, float* __restrict__ agg, int E) {
    long long tid = (long long)blockIdx.x * blockDim.x + threadIdx.x;
    int e = (int)(tid >> 5);
    if (e >= E) return;
    int f = ((int)tid & 31) * 4;
    int s = src[e], d = dst[e];
    float4 v = *(const float4*)(xs + (size_t)s * HID + f);
    float* p = agg + (size_t)d * HID + f;
    atomicAdd(p + 0, v.x);
    atomicAdd(p + 1, v.y);
    atomicAdd(p + 2, v.z);
    atomicAdd(p + 3, v.w);
}

// out = maybe_relu(agg * dinv[row] + bias[col])
__global__ void k_finalize(const float* __restrict__ agg, const float* __restrict__ dinv,
                           const float* __restrict__ bias, float* __restrict__ out,
                           long long total, int relu) {
    long long i = (long long)blockIdx.x * blockDim.x + threadIdx.x;
    if (i >= total) return;
    float v = agg[i] * dinv[i >> 7] + bias[i & 127];
    if (relu) v = fmaxf(v, 0.0f);
    out[i] = v;
}

// ---------------------------------------------------------------------------
// GRU elementwise
// ---------------------------------------------------------------------------
__device__ __forceinline__ float sigmoidf(float x) { return 1.0f / (1.0f + expf(-x)); }

// zr <- u = sigmoid(zr + bz);  rr <- sigmoid(rr + br) * hidden
__global__ void k_gates(float* __restrict__ zr, float* __restrict__ rr,
                        const float* __restrict__ hidden,
                        const float* __restrict__ bz, const float* __restrict__ br,
                        long long total) {
    long long i = (long long)blockIdx.x * blockDim.x + threadIdx.x;
    if (i >= total) return;
    int c = (int)(i & 127);
    zr[i] = sigmoidf(zr[i] + bz[c]);
    rr[i] = sigmoidf(rr[i] + br[c]) * hidden[i];
}

// cpre <- (1-u)*hidden + u*tanh(cpre + bc)
__global__ void k_gru_final(float* __restrict__ cpre, const float* __restrict__ u,
                            const float* __restrict__ hidden, const float* __restrict__ bc,
                            long long total) {
    long long i = (long long)blockIdx.x * blockDim.x + threadIdx.x;
    if (i >= total) return;
    float uu = u[i];
    float cand = tanhf(cpre[i] + bc[i & 127]);
    cpre[i] = (1.0f - uu) * hidden[i] + uu * cand;
}

// ---------------------------------------------------------------------------
extern "C" void kernel_launch(void* const* d_in, const int* in_sizes, int n_in,
                              void* d_out, int out_size, void* d_ws, size_t ws_size,
                              hipStream_t stream) {
    const float* x      = (const float*)d_in[0];
    const float* hidden = (const float*)d_in[1];
    const int*   ei     = (const int*)d_in[2];
    const float* W_in   = (const float*)d_in[3];
    const float* b_in   = (const float*)d_in[4];
    const float* W_hid  = (const float*)d_in[5];
    const float* b_hid  = (const float*)d_in[6];
    const float* W_z    = (const float*)d_in[7];
    const float* b_z    = (const float*)d_in[8];
    const float* W_r    = (const float*)d_in[9];
    const float* b_r    = (const float*)d_in[10];
    const float* W_c    = (const float*)d_in[11];
    const float* b_c    = (const float*)d_in[12];

    const int N = in_sizes[0] / HID;
    const int E = in_sizes[2] / 2;
    const int* src = ei;
    const int* dst = ei + E;

    // --- workspace layout ---
    char* w = (char*)d_ws;
    auto alignup = [](size_t v) { return (v + 255) & ~(size_t)255; };
    size_t NB = (size_t)N * HID * sizeof(float);
    size_t off = 0;
    float* dinv = (float*)(w + off); off = alignup(off + (size_t)N * sizeof(float));
    float* buf0 = (float*)(w + off); off = alignup(off + NB);
    float* buf1 = (float*)(w + off); off = alignup(off + NB);
    float* buf2 = (float*)(w + off); off = alignup(off + NB);
    __bf16* pWin  = (__bf16*)(w + off); off = alignup(off + (size_t)128 * HID * 2);
    __bf16* pWhid = (__bf16*)(w + off); off = alignup(off + (size_t)128 * HID * 2);
    __bf16* pWz   = (__bf16*)(w + off); off = alignup(off + (size_t)256 * HID * 2);
    __bf16* pWr   = (__bf16*)(w + off); off = alignup(off + (size_t)256 * HID * 2);
    __bf16* pWc   = (__bf16*)(w + off); off = alignup(off + (size_t)256 * HID * 2);
    (void)ws_size; (void)n_in; (void)out_size;

    float* out = (float*)d_out;

    const long long total = (long long)N * HID;
    const int gN  = (N + 255) / 256;
    const int gE  = (E + 255) / 256;
    const int gT  = (int)((total + 255) / 256);
    const int gS  = (int)(((long long)E * 32 + 255) / 256);
    const int gG  = (N + 255) / 256;                 // GEMM blocks (256 rows/block)
    const int gP1 = (128 * HID + 255) / 256;         // pack grids
    const int gP2 = (256 * HID + 255) / 256;

    // 1) normalization coefficients
    k_init_deg <<<gN, 256, 0, stream>>>(dinv, N);
    k_deg_edges<<<gE, 256, 0, stream>>>(dst, dinv, E);
    k_deg_rsqrt<<<gN, 256, 0, stream>>>(dinv, N);

    // 2) pack weights to bf16 WMMA fragments
    k_pack_w<<<gP1, 256, 0, stream>>>(W_in,  pWin, 128);
    k_pack_w<<<gP1, 256, 0, stream>>>(W_hid, pWhid, 128);
    k_pack_w<<<gP2, 256, 0, stream>>>(W_z,   pWz, 256);
    k_pack_w<<<gP2, 256, 0, stream>>>(W_r,   pWr, 256);
    k_pack_w<<<gP2, 256, 0, stream>>>(W_c,   pWc, 256);

    // 3) GCN layer 1:  h1 = relu(Agg(x @ W_in) + b_in)
    k_gemm_bf16 <<<gG, 256, 0, stream>>>(x, nullptr, pWin, buf0, N, 128);
    k_scale_init<<<gT, 256, 0, stream>>>(buf0, dinv, buf1, buf2, total);
    k_scatter   <<<gS, 256, 0, stream>>>(src, dst, buf1, buf2, E);
    k_finalize  <<<gT, 256, 0, stream>>>(buf2, dinv, b_in, buf0, total, 1);   // h1 -> buf0

    // 4) GCN layer 2:  emb = Agg(h1 @ W_hid) + b_hid
    k_gemm_bf16 <<<gG, 256, 0, stream>>>(buf0, nullptr, pWhid, buf1, N, 128);
    k_scale_init<<<gT, 256, 0, stream>>>(buf1, dinv, buf1, buf2, total);      // xs in place
    k_scatter   <<<gS, 256, 0, stream>>>(src, dst, buf1, buf2, E);
    k_finalize  <<<gT, 256, 0, stream>>>(buf2, dinv, b_hid, buf0, total, 0);  // emb -> buf0

    // 5) GRU gates:  u, r  from [emb | hidden]
    k_gemm_bf16<<<gG, 256, 0, stream>>>(buf0, hidden, pWz, buf1, N, 256);     // zpre
    k_gemm_bf16<<<gG, 256, 0, stream>>>(buf0, hidden, pWr, buf2, N, 256);     // rpre
    k_gates    <<<gT, 256, 0, stream>>>(buf1, buf2, hidden, b_z, b_r, total); // buf1=u, buf2=r*h

    // 6) candidate + blend:  out = (1-u)*h + u*tanh([emb | r*h] @ W_c + b_c)
    k_gemm_bf16<<<gG, 256, 0, stream>>>(buf0, buf2, pWc, out, N, 256);        // cpre -> d_out
    k_gru_final<<<gT, 256, 0, stream>>>(out, buf1, hidden, b_c, total);
}
